// SelfAttention_60060822667534
// MI455X (gfx1250) — matrix operand order repaired and checked by reading, not compiled, not run on hardware
//
#include <hip/hip_runtime.h>
#include <hip/hip_bf16.h>

// MI455X / gfx1250, wave32. bf16 WMMA 16x16x32 with f32 accumulation.
//
// Problem: B=8, H=W=64, C=256, C8=32, HW=4096.
// Raw-reshape index algebra (per batch):
//   qf[n, j] = q[P=j*128 + n/32][d=n%32]
//   kf[j, m] = k[P=j*128 + m/32][d=m%32]
//   vf[c, n] = v[P=c*16  + n/256][cc=n%256]
//   result[p, cc] = gamma * out_f[p/16][(p%16)*256 + cc] + x[p, cc]
//
// Pipeline:
//   prep_kernel : pack Wq|Wk|Wv -> Wall[256][320] bf16, biases -> [320] f32
//   proj_kernel : WMMA GEMM x(fp32->bf16 in-reg) @ Wall, scatter to:
//                   Qf [B][4096 n][32 j], Kf [B][32 j][4096 m],
//                   vb [B][4096 p][256 cc]
//   attn_kernel : fused sigmoid(Q@K) -> double-buffered LDS -> V@attn,
//                 V operand register-pipelined across the barrier,
//                 epilogue gamma*out + x. Out tile 256c x 64m per workgroup.

typedef __attribute__((ext_vector_type(16))) __bf16 v16bf;
typedef __attribute__((ext_vector_type(8)))  __bf16 v8bf;
typedef __attribute__((ext_vector_type(8)))  float  v8f;

#define BATCH 8
#define HW    4096
#define CCH   256
#define NCOL  320   // 32 (q) + 32 (k) + 256 (v)
#define APAD  72    // attn LDS row stride in bf16 (144B = 16B-aligned)

__device__ __forceinline__ v16bf cat8(v8bf lo, v8bf hi8) {
  v16bf r;
#pragma unroll
  for (int t = 0; t < 8; ++t) { r[t] = lo[t]; r[t + 8] = hi8[t]; }
  return r;
}

__device__ __forceinline__ v16bf cvt16(v8f lo, v8f hi8) {
  v16bf r;
#pragma unroll
  for (int t = 0; t < 8; ++t) { r[t] = (__bf16)lo[t]; r[t + 8] = (__bf16)hi8[t]; }
  return r;
}

// ---------------------------------------------------------------------------
// Kernel 0: weight packing (tiny: 320 blocks)
// ---------------------------------------------------------------------------
__global__ __launch_bounds__(256) void prep_kernel(
    const float* __restrict__ Wq, const float* __restrict__ bq,
    const float* __restrict__ Wk, const float* __restrict__ bk,
    const float* __restrict__ Wv, const float* __restrict__ bv,
    __bf16* __restrict__ Wall, float* __restrict__ bias) {
  const int idx = blockIdx.x * 256 + threadIdx.x;   // [0, 256*320)
  const int c = idx / NCOL, o = idx % NCOL;
  float w = (o < 32) ? Wq[c * 32 + o]
          : (o < 64) ? Wk[c * 32 + (o - 32)]
                     : Wv[c * 256 + (o - 64)];
  Wall[idx] = (__bf16)w;
  if (idx < NCOL)
    bias[idx] = (idx < 32) ? bq[idx] : (idx < 64) ? bk[idx - 32] : bv[idx - 64];
}

// ---------------------------------------------------------------------------
// Kernel 1: q/k/v projections, one WMMA GEMM [4096 x 256] @ [256 x 320].
// Workgroup: 8 waves, 32 pixels x 320 outputs; x converted fp32->bf16 in-reg.
// ---------------------------------------------------------------------------
__global__ __launch_bounds__(256) void proj_kernel(
    const float* __restrict__ x, const __bf16* __restrict__ Wall,
    const float* __restrict__ bias, __bf16* __restrict__ Qf,
    __bf16* __restrict__ Kf, __bf16* __restrict__ vb) {
  const int b     = blockIdx.y;
  const int p0    = blockIdx.x * 32;
  const int wave  = threadIdx.x >> 5;
  const int lane  = threadIdx.x & 31;
  const int lm    = lane & 15;
  const int hi    = (lane >= 16) ? 8 : 0;   // A-layout K-half split
  const int phalf = wave & 1;
  const int ct0   = wave >> 1;

  const float* xrow =
      x + ((size_t)b * HW + (size_t)(p0 + phalf * 16 + lm)) * CCH;

  v8f acc[5];
#pragma unroll
  for (int u = 0; u < 5; ++u)
    acc[u] = (v8f){0.f, 0.f, 0.f, 0.f, 0.f, 0.f, 0.f, 0.f};

  for (int kc = 0; kc < 8; ++kc) {
    // A: x tile [16 px x 32 ch], fp32 loads (32B-aligned runs) -> bf16.
    v8f a0 = *(const v8f*)(xrow + kc * 32 + hi);
    v8f a1 = *(const v8f*)(xrow + kc * 32 + hi + 16);
    v16bf amat = cvt16(a0, a1);
    // B: Wall chunk [32 ch x 16 outs]; lane = N col, 16 K rows (table order).
    const __bf16* wcol = Wall + (size_t)(kc * 32) * NCOL + lm;
#pragma unroll
    for (int u = 0; u < 5; ++u) {
      const int ct = ct0 + u * 4;
      v16bf bmat;
#pragma unroll
      for (int t = 0; t < 8; ++t) {
        bmat[t]     = wcol[(size_t)(hi + t) * NCOL + ct * 16];
        bmat[t + 8] = wcol[(size_t)(hi + 16 + t) * NCOL + ct * 16];
      }
      acc[u] = __builtin_amdgcn_wmma_f32_16x16x32_bf16(
          false, amat, false, bmat, (short)0, acc[u], false, false);
    }
  }

  // Bias add, bf16 convert, scatter into WMMA-operand layouts.
#pragma unroll
  for (int u = 0; u < 5; ++u) {
    const int ct  = ct0 + u * 4;
    const int col = ct * 16 + lm;          // N of C/D layout
    const float bsv = bias[col];
#pragma unroll
    for (int i = 0; i < 8; ++i) {
      const int P = p0 + phalf * 16 + i + hi;  // M of C/D layout
      const __bf16 v = (__bf16)(acc[u][i] + bsv);
      const int j = P >> 7;
      const int r = P & 127;
      if (col < 32) {                      // q -> Qf[n][j]
        Qf[((size_t)b * HW + (size_t)(r * 32 + col)) * 32 + j] = v;
      } else if (col < 64) {               // k -> Kf[j][m]
        Kf[((size_t)b * 32 + j) * HW + (size_t)(r * 32 + (col - 32))] = v;
      } else {                             // v -> vb[P][cc]
        vb[((size_t)b * HW + P) * CCH + (col - 64)] = v;
      }
    }
  }
}

// ---------------------------------------------------------------------------
// Kernel 2: fused out_f[256, 64m] = vf @ sigmoid(qf @ kf).
// Grid (64 m-tiles, 8 batches), 8 waves. Per n-chunk (32 wide):
//   produce : each wave computes ONE 16x16 logits tile of chunk it+1
//             (Kf B-operand hoisted: loop-invariant), sigmoid -> LDS buf^1
//   prefetch: V A-tiles for chunk it+1 issued into shadow registers
//             (no LDS/barrier dependence -> loads fly across the barrier)
//   consume : 8 WMMAs (2 c-tiles x 4 m-tiles) on resident regs + LDS buf
//   one __syncthreads per chunk.
// ---------------------------------------------------------------------------
__global__ __launch_bounds__(256) void attn_kernel(
    const __bf16* __restrict__ Qf, const __bf16* __restrict__ Kf,
    const __bf16* __restrict__ vb, const float* __restrict__ x,
    const float* __restrict__ gamma, float* __restrict__ out) {
  __shared__ __bf16 attn_s[2][32][APAD];   // [buf][n_loc][m_loc]

  const int b    = blockIdx.y;
  const int m0   = blockIdx.x * 64;
  const int wave = threadIdx.x >> 5;
  const int lane = threadIdx.x & 31;
  const int lm   = lane & 15;
  const int hi   = (lane >= 16) ? 8 : 0;
  const int cbase = wave * 32;
  const int ltn  = wave >> 2;              // logits row-tile (0..1)
  const int ltm  = wave & 3;               // logits col-tile (0..3)

  const __bf16* Qb = Qf + (size_t)b * HW * 32;
  const __bf16* Kb = Kf + (size_t)b * 32 * HW;
  const __bf16* Vb = vb + (size_t)b * HW * CCH;

  // Loop-invariant logits B operand: Kf col m0+ltm*16+lm, 16 j rows (table order).
  const __bf16* kcol = Kb + m0 + ltm * 16 + lm;
  v16bf bmatK;
#pragma unroll
  for (int t = 0; t < 8; ++t) {
    bmatK[t]     = kcol[(size_t)(hi + t) * HW];
    bmatK[t + 8] = kcol[(size_t)(hi + 16 + t) * HW];
  }

  v8f acc[2][4];
#pragma unroll
  for (int tn = 0; tn < 2; ++tn)
#pragma unroll
    for (int tm = 0; tm < 4; ++tm)
      acc[tn][tm] = (v8f){0.f, 0.f, 0.f, 0.f, 0.f, 0.f, 0.f, 0.f};

  // Preload V A-tiles for chunk 0 (n0=0 -> pr=0, cc0=0).
  v16bf amatV[2];
#pragma unroll
  for (int tn = 0; tn < 2; ++tn) {
    const __bf16* vrow = Vb + (size_t)(cbase + tn * 16 + lm) * 16 * CCH;
    amatV[tn] = cat8(*(const v8bf*)(vrow + hi),
                     *(const v8bf*)(vrow + hi + 16));
  }

  // Prologue: produce chunk 0 into buffer 0.
  {
    const __bf16* qrow = Qb + (size_t)(ltn * 16 + lm) * 32;
    v16bf amat = cat8(*(const v8bf*)(qrow + hi),
                      *(const v8bf*)(qrow + hi + 16));
    v8f lg = (v8f){0.f, 0.f, 0.f, 0.f, 0.f, 0.f, 0.f, 0.f};
    lg = __builtin_amdgcn_wmma_f32_16x16x32_bf16(
        false, amat, false, bmatK, (short)0, lg, false, false);
#pragma unroll
    for (int i = 0; i < 8; ++i) {
      const float s = 1.0f / (1.0f + __expf(-lg[i]));
      attn_s[0][ltn * 16 + i + hi][ltm * 16 + lm] = (__bf16)s;
    }
  }
  __syncthreads();

#pragma unroll 2
  for (int it = 0; it < 128; ++it) {
    const int cur = it & 1;

    // Produce logits of chunk it+1 into the other buffer.
    if (it < 127) {
      const int n1 = (it + 1) * 32;
      const __bf16* qrow = Qb + (size_t)(n1 + ltn * 16 + lm) * 32;
      v16bf amat = cat8(*(const v8bf*)(qrow + hi),
                        *(const v8bf*)(qrow + hi + 16));
      v8f lg = (v8f){0.f, 0.f, 0.f, 0.f, 0.f, 0.f, 0.f, 0.f};
      lg = __builtin_amdgcn_wmma_f32_16x16x32_bf16(
          false, amat, false, bmatK, (short)0, lg, false, false);
#pragma unroll
      for (int i = 0; i < 8; ++i) {
        const float s = 1.0f / (1.0f + __expf(-lg[i]));
        attn_s[cur ^ 1][ltn * 16 + i + hi][ltm * 16 + lm] = (__bf16)s;
      }
    }

    // Prefetch V A-tiles for the next chunk (stays in flight across barrier).
    v16bf nextV[2];
    {
      const int n1  = (it < 127) ? (it + 1) * 32 : it * 32;
      const int cc1 = n1 & 255;
      const int pr1 = n1 >> 8;
#pragma unroll
      for (int tn = 0; tn < 2; ++tn) {
        const __bf16* vrow =
            Vb + ((size_t)(cbase + tn * 16 + lm) * 16 + pr1) * CCH + cc1;
        nextV[tn] = cat8(*(const v8bf*)(vrow + hi),
                         *(const v8bf*)(vrow + hi + 16));
      }
    }

    // Consume chunk it: resident V registers x LDS attn tile.
#pragma unroll
    for (int tn = 0; tn < 2; ++tn) {
#pragma unroll
      for (int tm = 0; tm < 4; ++tm) {
        v16bf bmat;
#pragma unroll
        for (int t = 0; t < 8; ++t) {
          bmat[t]     = attn_s[cur][hi + t][tm * 16 + lm];
          bmat[t + 8] = attn_s[cur][hi + 16 + t][tm * 16 + lm];
        }
        acc[tn][tm] = __builtin_amdgcn_wmma_f32_16x16x32_bf16(
            false, amatV[tn], false, bmat, (short)0, acc[tn][tm], false, false);
      }
    }
    amatV[0] = nextV[0];
    amatV[1] = nextV[1];
    __syncthreads();
  }

  // Epilogue: inverse reshape + gamma*out_f + x (coalesced 16-float runs).
  const float g  = gamma[0];
  const int pcol = m0 >> 8;
  const int ccb  = m0 & 255;
#pragma unroll
  for (int tn = 0; tn < 2; ++tn)
#pragma unroll
    for (int tm = 0; tm < 4; ++tm)
#pragma unroll
      for (int i = 0; i < 8; ++i) {
        const int c  = cbase + tn * 16 + i + hi;
        const int p  = c * 16 + pcol;
        const int cc = ccb + tm * 16 + lm;
        const size_t idx = ((size_t)b * HW + p) * CCH + cc;
        out[idx] = g * acc[tn][tm][i] + x[idx];
      }
}

// ---------------------------------------------------------------------------
extern "C" void kernel_launch(void* const* d_in, const int* in_sizes, int n_in,
                              void* d_out, int out_size, void* d_ws,
                              size_t ws_size, hipStream_t stream) {
  const float* x     = (const float*)d_in[0];
  const float* Wq    = (const float*)d_in[1];
  const float* bq    = (const float*)d_in[2];
  const float* Wk    = (const float*)d_in[3];
  const float* bk    = (const float*)d_in[4];
  const float* Wv    = (const float*)d_in[5];
  const float* bv    = (const float*)d_in[6];
  const float* gamma = (const float*)d_in[7];
  float* out = (float*)d_out;

  // Workspace layout (~20.3 MB):
  char* ws = (char*)d_ws;
  __bf16* vb   = (__bf16*)(ws);                 // 16 MB : v projection
  __bf16* Qf   = (__bf16*)(ws + 16777216);      //  2 MB : qf n-major
  __bf16* Kf   = (__bf16*)(ws + 18874368);      //  2 MB : kf j-major
  __bf16* Wall = (__bf16*)(ws + 20971520);      // 160 KB: packed weights
  float*  bias = (float*) (ws + 21135360);      // 1.25 KB packed biases

  prep_kernel<<<dim3((256 * NCOL) / 256), dim3(256), 0, stream>>>(
      Wq, bq, Wk, bk, Wv, bv, Wall, bias);
  proj_kernel<<<dim3(HW / 32, BATCH), dim3(256), 0, stream>>>(
      x, Wall, bias, Qf, Kf, vb);
  attn_kernel<<<dim3(HW / 64, BATCH), dim3(256), 0, stream>>>(
      Qf, Kf, vb, x, gamma, out);
}
